// RoPEMultiHeadAttentionWithGQA_31636729102512
// MI455X (gfx1250) — matrix operand order repaired and checked by reading, not compiled, not run on hardware
//
#include <hip/hip_runtime.h>

typedef __attribute__((ext_vector_type(16))) __bf16 v16bf;
typedef __attribute__((ext_vector_type(8)))  float  v8f;

#define DEVINL __device__ __forceinline__

namespace {
constexpr int kB = 2, kS = 2048, kD = 2048, kHQ = 16, kHKV = 4, kHD = 128;
constexpr int kG = kHQ / kHKV;                 // 4 q-heads per kv-head
constexpr float kScale = 0.08838834764831845f; // 1/sqrt(128)
}

// ---------------- helpers ----------------

DEVINL __bf16 f2bf(float f) {
  unsigned u = __builtin_bit_cast(unsigned, f);
  unsigned r = u + 0x7FFFu + ((u >> 16) & 1u);  // round-to-nearest-even
  unsigned short h = (unsigned short)(r >> 16);
  return __builtin_bit_cast(__bf16, h);
}

DEVINL v8f wmma_bf16(v16bf a, v16bf b, v8f c) {
  // D(f32 16x16) = A(bf16 16x32) * B(bf16 32x16) + C
  return __builtin_amdgcn_wmma_f32_16x16x32_bf16(false, a, false, b, (short)0, c,
                                                 false, false);
}

// K index held by element e of a 16-bit A fragment (16x32), matching B frags.
// e=0..15 -> K = base+0..15 (contiguous), where base = hi ? 16 : 0.
// So a fragment is one contiguous 32-byte run.
DEVINL int a_k(int e, int hi) {
  return (hi ? 16 : 0) + e;
}

DEVINL float grp_max16(float v) {
#pragma unroll
  for (int o = 1; o < 16; o <<= 1) v = fmaxf(v, __shfl_xor(v, o, 16));
  return v;
}
DEVINL float grp_sum16(float v) {
#pragma unroll
  for (int o = 1; o < 16; o <<= 1) v += __shfl_xor(v, o, 16);
  return v;
}
DEVINL float wave_sum32(float v) {
#pragma unroll
  for (int o = 1; o < 32; o <<= 1) v += __shfl_xor(v, o, 32);
  return v;
}

// ---------------- pack kernels ----------------

__global__ void pack_bf16_kernel(const float* __restrict__ src,
                                 __bf16* __restrict__ dst, size_t n) {
  size_t i = (size_t)blockIdx.x * blockDim.x + threadIdx.x;
  if (i < n) dst[i] = f2bf(src[i]);
}

// src: K x N (f32, row-major) -> dst: N x K (bf16, row-major) == B^T
__global__ void pack_bf16_t_kernel(const float* __restrict__ src,
                                   __bf16* __restrict__ dst, int K, int N) {
  size_t i = (size_t)blockIdx.x * blockDim.x + threadIdx.x;
  if (i >= (size_t)K * N) return;
  int n = (int)(i % N);
  int k = (int)(i / N);
  dst[(size_t)n * K + k] = f2bf(src[i]);
}

// ---------------- GEMM (cache-direct, TN form) ----------------
// C[M][N](f32) = A[M][K](bf16, row-major) * Bt[N][K](bf16, row-major == B^T).
// One wave computes a 32x64 tile; all fragment loads are contiguous 16B/32B
// runs -> global_load_b128.  Weight panels live in the 192MB L2.
__global__ __launch_bounds__(128) void gemm_bf16_tn_kernel(
    const __bf16* __restrict__ A, const __bf16* __restrict__ Bt,
    float* __restrict__ C, int M, int N, int K) {
  const int lane = threadIdx.x & 31;
  const int lm = lane & 15;
  const int hi = lane >> 4;
  const int wid = (int)((blockIdx.x * blockDim.x + threadIdx.x) >> 5);
  const int nTiles = N >> 6;
  const int totalWaves = (M >> 5) * nTiles;
  if (wid >= totalWaves) return;
  const int m0 = (wid / nTiles) * 32;
  const int n0 = (wid % nTiles) * 64;

  const __bf16* arow0 = A + (size_t)(m0 + lm) * K;
  const __bf16* arow1 = A + (size_t)(m0 + 16 + lm) * K;

  v8f zero = {};
  v8f acc[2][4];
#pragma unroll
  for (int i = 0; i < 2; ++i)
#pragma unroll
    for (int j = 0; j < 4; ++j) acc[i][j] = zero;

  for (int k0 = 0; k0 < K; k0 += 32) {
    if (k0 + 128 < K) {
      __builtin_prefetch(arow0 + k0 + 128, 0, 3);
      __builtin_prefetch(arow1 + k0 + 128, 0, 3);
    }
    v16bf a0, a1;
#pragma unroll
    for (int e = 0; e < 16; ++e) {
      int kk = k0 + a_k(e, hi);
      a0[e] = arow0[kk];
      a1[e] = arow1[kk];
    }
#pragma unroll
    for (int nf = 0; nf < 4; ++nf) {
      const __bf16* brow = Bt + (size_t)(n0 + nf * 16 + lm) * K + k0 + hi * 16;
      v16bf bfrg;  // lane: column n fixed, 16 consecutive k -> one 32B run
#pragma unroll
      for (int e = 0; e < 16; ++e) bfrg[e] = brow[e];
      acc[0][nf] = wmma_bf16(a0, bfrg, acc[0][nf]);
      acc[1][nf] = wmma_bf16(a1, bfrg, acc[1][nf]);
    }
  }

#pragma unroll
  for (int half = 0; half < 2; ++half)
#pragma unroll
    for (int nf = 0; nf < 4; ++nf)
#pragma unroll
      for (int r = 0; r < 8; ++r)
        C[(size_t)(m0 + half * 16 + r + hi * 8) * N + n0 + nf * 16 + lm] =
            acc[half][nf][r];
}

// ---------------- RMSNorm + RoPE ----------------
// One wave per (b, s, head): RMSNorm(128) + RoPE, f32 in -> bf16 out,
// relayout to [b][h][s][128].
__global__ void qk_norm_rope_kernel(const float* __restrict__ src,
                                    __bf16* __restrict__ dst,
                                    const float* __restrict__ nw,
                                    const float* __restrict__ cost,
                                    const float* __restrict__ sint, int H) {
  const int wid = (int)((blockIdx.x * blockDim.x + threadIdx.x) >> 5);
  const int lane = threadIdx.x & 31;
  const int total = kB * kS * H;
  if (wid >= total) return;
  const int h = wid % H;
  const int s = (wid / H) % kS;
  const int b = wid / (H * kS);
  const float* row = src + ((size_t)(b * kS + s) * H + h) * kHD;
  const float* cr = cost + (size_t)s * kHD;
  const float* sr = sint + (size_t)s * kHD;
  const int d0 = lane, d1 = lane + 32, d2 = lane + 64, d3 = lane + 96;

  float v0 = row[d0], v1 = row[d1], v2 = row[d2], v3 = row[d3];
  float ssq = wave_sum32(v0 * v0 + v1 * v1 + v2 * v2 + v3 * v3);
  float inv = rsqrtf(ssq * (1.0f / kHD) + 1e-8f);
  v0 *= inv * (1.0f + nw[d0]);
  v1 *= inv * (1.0f + nw[d1]);
  v2 *= inv * (1.0f + nw[d2]);
  v3 *= inv * (1.0f + nw[d3]);
  // rope: pairs (d, d+64)
  float y0 = v0 * cr[d0] - v2 * sr[d0];
  float y1 = v1 * cr[d1] - v3 * sr[d1];
  float y2 = v2 * cr[d2] + v0 * sr[d2];
  float y3 = v3 * cr[d3] + v1 * sr[d3];

  __bf16* out = dst + ((size_t)(b * H + h) * kS + s) * kHD;
  out[d0] = f2bf(y0);
  out[d1] = f2bf(y1);
  out[d2] = f2bf(y2);
  out[d3] = f2bf(y3);
}

// V f32 [b][s][kvh][128] -> bf16 Vt [b][kvh][128][S] (d-major for PV B-frags)
__global__ void v_pack_transpose_kernel(const float* __restrict__ src,
                                        __bf16* __restrict__ dst) {
  size_t i = (size_t)blockIdx.x * blockDim.x + threadIdx.x;
  size_t n = (size_t)kB * kS * kHKV * kHD;
  if (i >= n) return;
  int d = (int)(i % kHD);
  int h = (int)((i / kHD) % kHKV);
  int s = (int)((i / ((size_t)kHD * kHKV)) % kS);
  int b = (int)(i / ((size_t)kHD * kHKV * kS));
  dst[((size_t)(b * kHKV + h) * kHD + d) * kS + s] = f2bf(src[i]);
}

// ---------------- attention ----------------
// Flash-style sliding-window causal GQA attention. One wave = 16 query rows of
// one (b, q-head). Q@K^T and P@V via bf16 WMMA, online softmax in f32.
// All global fragment loads are contiguous; only the P C-layout -> A-layout
// relayout round-trips through LDS.
__global__ __launch_bounds__(128) void attn_kernel(
    const __bf16* __restrict__ Qr, const __bf16* __restrict__ Kr,
    const __bf16* __restrict__ Vt, const unsigned char* __restrict__ pad,
    const int* __restrict__ winp, __bf16* __restrict__ Ob) {
  __shared__ __bf16 pb[4][16][32];  // per-wave P staging
  const int w = threadIdx.x >> 5;
  const int lane = threadIdx.x & 31;
  const int lm = lane & 15;
  const int hi = lane >> 4;
  const int wg = blockIdx.x * 4 + w;
  const int qt = wg & (kS / 16 - 1);        // 128 query tiles
  const int qh = (wg >> 7) & (kHQ - 1);     // 16 q-heads
  const int b = wg >> 11;
  const int q0 = qt * 16;
  const int kvh = qh / kG;
  const int win = winp[0];
  const float NEG_INF = -__builtin_inff();

  const __bf16* Qb = Qr + (size_t)(b * kHQ + qh) * kS * kHD;
  const __bf16* Kb = Kr + (size_t)(b * kHKV + kvh) * kS * kHD;
  const __bf16* Vb = Vt + (size_t)(b * kHKV + kvh) * kHD * kS;

  // Q fragments for the whole tile: 4 K-slices of 32 over HD=128
  v16bf qa[4];
#pragma unroll
  for (int kk = 0; kk < 4; ++kk)
#pragma unroll
    for (int e = 0; e < 16; ++e)
      qa[kk][e] = Qb[(size_t)(q0 + lm) * kHD + kk * 32 + a_k(e, hi)];

  v8f zero = {};
  v8f acc[8];  // 8 n-fragments cover HD=128 output columns
  float rowmax[8], rowsum[8];
  bool padq[8];
#pragma unroll
  for (int f = 0; f < 8; ++f) acc[f] = zero;
#pragma unroll
  for (int r = 0; r < 8; ++r) {
    rowmax[r] = NEG_INF;
    rowsum[r] = 0.f;
    padq[r] = pad[(size_t)b * kS + q0 + r + hi * 8] != 0;
  }

  int jlo = q0 - win;
  if (jlo < 0) jlo = 0;
  jlo &= ~31;
  for (int j0 = jlo; j0 <= q0 + 15; j0 += 32) {
    if (j0 + 32 <= q0 + 15) {
      __builtin_prefetch(Kb + (size_t)(j0 + 32 + lm) * kHD, 0, 3);
      __builtin_prefetch(Kb + (size_t)(j0 + 48 + lm) * kHD, 0, 3);
    }
    // ---- scores S = Q @ K^T (two 16-key n-tiles) ----
    v8f sc0 = zero, sc1 = zero;
#pragma unroll
    for (int kk = 0; kk < 4; ++kk) {
      const __bf16* kr0 = Kb + (size_t)(j0 + lm) * kHD + kk * 32 + hi * 16;
      const __bf16* kr1 = Kb + (size_t)(j0 + 16 + lm) * kHD + kk * 32 + hi * 16;
      v16bf kb0, kb1;  // B frag: n = key (lm), 16 consecutive d per lane
#pragma unroll
      for (int e = 0; e < 16; ++e) {
        kb0[e] = kr0[e];
        kb1[e] = kr1[e];
      }
      sc0 = wmma_bf16(qa[kk], kb0, sc0);
      sc1 = wmma_bf16(qa[kk], kb1, sc1);
    }

    const bool pk0 = pad[(size_t)b * kS + j0 + lm] != 0;
    const bool pk1 = pad[(size_t)b * kS + j0 + 16 + lm] != 0;

    // ---- online softmax (a row lives in one 16-lane group) ----
#pragma unroll
    for (int r = 0; r < 8; ++r) {
      const int qi = q0 + r + hi * 8;
      const int k0i = j0 + lm, k1i = j0 + 16 + lm;
      float s0 = sc0[r] * kScale;
      float s1 = sc1[r] * kScale;
      if (qi < k0i || qi - k0i > win || padq[r] || pk0) s0 = NEG_INF;
      if (qi < k1i || qi - k1i > win || padq[r] || pk1) s1 = NEG_INF;
      float nm = fmaxf(rowmax[r], grp_max16(fmaxf(s0, s1)));
      float p0 = 0.f, p1 = 0.f, fct = 1.f;
      if (nm > NEG_INF) {
        fct = __expf(rowmax[r] - nm);  // exp(-inf)=0 when previous max was -inf
        p0 = __expf(s0 - nm);
        p1 = __expf(s1 - nm);
      }
      rowmax[r] = nm;
      rowsum[r] = rowsum[r] * fct + grp_sum16(p0 + p1);
#pragma unroll
      for (int f = 0; f < 8; ++f) acc[f][r] *= fct;
      pb[w][r + hi * 8][lm] = f2bf(p0);
      pb[w][r + hi * 8][lm + 16] = f2bf(p1);
    }

    // ---- out += P @ V ----
    v16bf pf;  // A frag from LDS: m = lm, kdim = key 0..31 (one 32B run per lane)
#pragma unroll
    for (int e = 0; e < 16; ++e) pf[e] = pb[w][lm][a_k(e, hi)];
#pragma unroll
    for (int f = 0; f < 8; ++f) {
      const __bf16* vr = Vb + (size_t)(f * 16 + lm) * kS + j0 + hi * 16;
      v16bf vf;  // B frag: n = d (lm), 16 consecutive keys per lane (Vt layout)
#pragma unroll
      for (int e = 0; e < 16; ++e) vf[e] = vr[e];
      acc[f] = wmma_bf16(pf, vf, acc[f]);
    }
  }

  // ---- normalize + store bf16 attn output [b][s][HQ*HD] ----
#pragma unroll
  for (int r = 0; r < 8; ++r) {
    float rinv = 1.0f / rowsum[r];
    const int row = q0 + r + hi * 8;
#pragma unroll
    for (int f = 0; f < 8; ++f)
      Ob[((size_t)(b * kS + row) * kHQ + qh) * kHD + f * 16 + lm] =
          f2bf(acc[f][r] * rinv);
  }
}

// ---------------- launch ----------------

extern "C" void kernel_launch(void* const* d_in, const int* in_sizes, int n_in,
                              void* d_out, int out_size, void* d_ws,
                              size_t ws_size, hipStream_t stream) {
  (void)in_sizes; (void)n_in; (void)out_size; (void)ws_size;
  const float* x   = (const float*)d_in[0];
  const float* Wq  = (const float*)d_in[1];
  const float* Wk  = (const float*)d_in[2];
  const float* Wv  = (const float*)d_in[3];
  const float* Wo  = (const float*)d_in[4];
  const float* qnw = (const float*)d_in[5];
  const float* knw = (const float*)d_in[6];
  const float* ct  = (const float*)d_in[7];
  const float* st  = (const float*)d_in[8];
  const unsigned char* pad = (const unsigned char*)d_in[9];
  const int* win = (const int*)d_in[10];

  char* ws = (char*)d_ws;
  size_t off = 0;
  auto alloc = [&](size_t bytes) -> void* {
    void* p = ws + off;
    off += (bytes + 255) & ~(size_t)255;
    return p;
  };
  const size_t nX  = (size_t)kB * kS * kD;
  const size_t nWq = (size_t)kD * kHQ * kHD;
  const size_t nWk = (size_t)kD * kHKV * kHD;
  const size_t nQ  = (size_t)kB * kS * kHQ * kHD;
  const size_t nK  = (size_t)kB * kS * kHKV * kHD;

  __bf16* xb  = (__bf16*)alloc(nX * 2);
  __bf16* wqt = (__bf16*)alloc(nWq * 2);  // transposed: [HQ*HD][D]
  __bf16* wkt = (__bf16*)alloc(nWk * 2);  // transposed: [HKV*HD][D]
  __bf16* wvt = (__bf16*)alloc(nWk * 2);
  __bf16* wot = (__bf16*)alloc(nWq * 2);  // transposed: [D][HQ*HD]
  float*  qf  = (float*)alloc(nQ * 4);
  float*  kf  = (float*)alloc(nK * 4);
  float*  vf  = (float*)alloc(nK * 4);
  __bf16* qrb = (__bf16*)alloc(nQ * 2);
  __bf16* krb = (__bf16*)alloc(nK * 2);
  __bf16* vtb = (__bf16*)alloc(nK * 2);
  __bf16* ob  = (__bf16*)alloc(nQ * 2);

  pack_bf16_kernel<<<dim3((unsigned)((nX + 255) / 256)), 256, 0, stream>>>(x, xb, nX);
  pack_bf16_t_kernel<<<dim3((unsigned)((nWq + 255) / 256)), 256, 0, stream>>>(
      Wq, wqt, kD, kHQ * kHD);
  pack_bf16_t_kernel<<<dim3((unsigned)((nWk + 255) / 256)), 256, 0, stream>>>(
      Wk, wkt, kD, kHKV * kHD);
  pack_bf16_t_kernel<<<dim3((unsigned)((nWk + 255) / 256)), 256, 0, stream>>>(
      Wv, wvt, kD, kHKV * kHD);
  pack_bf16_t_kernel<<<dim3((unsigned)((nWq + 255) / 256)), 256, 0, stream>>>(
      Wo, wot, kHQ * kHD, kD);

  // QKV projections (bf16 WMMA, f32 out).  waves = (M/32)*(N/64), 4 waves/block
  auto gemm_grid = [](int M, int N) {
    return dim3((unsigned)(((M >> 5) * (N >> 6) + 3) / 4));
  };
  gemm_bf16_tn_kernel<<<gemm_grid(kB * kS, kHQ * kHD), 128, 0, stream>>>(
      xb, wqt, qf, kB * kS, kHQ * kHD, kD);
  gemm_bf16_tn_kernel<<<gemm_grid(kB * kS, kHKV * kHD), 128, 0, stream>>>(
      xb, wkt, kf, kB * kS, kHKV * kHD, kD);
  gemm_bf16_tn_kernel<<<gemm_grid(kB * kS, kHKV * kHD), 128, 0, stream>>>(
      xb, wvt, vf, kB * kS, kHKV * kHD, kD);

  // RMSNorm + RoPE (Q, K) and V transpose/pack
  qk_norm_rope_kernel<<<dim3((kB * kS * kHQ) / 8), 256, 0, stream>>>(
      qf, qrb, qnw, ct, st, kHQ);
  qk_norm_rope_kernel<<<dim3((kB * kS * kHKV) / 8), 256, 0, stream>>>(
      kf, krb, knw, ct, st, kHKV);
  v_pack_transpose_kernel<<<dim3((unsigned)(nK / 256)), 256, 0, stream>>>(vf, vtb);

  // Attention: one wave per (b, q-head, 16-query tile)
  attn_kernel<<<dim3((kB * kHQ * (kS / 16)) / 4), 128, 0, stream>>>(
      qrb, krb, vtb, pad, win, ob);

  // Output projection -> f32 d_out
  gemm_bf16_tn_kernel<<<gemm_grid(kB * kS, kD), 128, 0, stream>>>(
      ob, wot, (float*)d_out, kB * kS, kD, kHQ * kHD);
}
